// scirex_bert_171798692628
// MI455X (gfx1250) — matrix operand order repaired and checked by reading, not compiled, not run on hardware
//
#include <hip/hip_runtime.h>

typedef __attribute__((ext_vector_type(16))) _Float16 v16h;
typedef __attribute__((ext_vector_type(8)))  float    v8f;

#define GN 8192          // nodes
#define GNW 256          // 32-bit mask words per row (8192/32)
#define GF 16            // features
#define GALPHA 0.2f      // LeakyReLU slope

// ---------------------------------------------------------------------------
// Kernel 1: pack adjacency (int32 0/1) + self loops into a bitmask.
// One thread per 32-bit output word; 8 x int4 (128-bit) loads per thread.
// This is the single 256MB HBM scan; both layers afterwards touch only 8MB
// of mask, which stays resident in the 192MB L2.
// ---------------------------------------------------------------------------
__global__ void pack_mask_kernel(const int* __restrict__ adj,
                                 unsigned* __restrict__ mask) {
  int g   = blockIdx.x * blockDim.x + threadIdx.x;   // word id in [0, GN*GNW)
  int row = g >> 8;                                  // /GNW
  int w   = g & (GNW - 1);
  const int4* p = (const int4*)(adj + (size_t)row * GN + (size_t)w * 32);
  unsigned bits = 0u;
#pragma unroll
  for (int q = 0; q < 8; ++q) {
    int4 v = p[q];
    bits |= (unsigned)(v.x != 0) << (q * 4 + 0);
    bits |= (unsigned)(v.y != 0) << (q * 4 + 1);
    bits |= (unsigned)(v.z != 0) << (q * 4 + 2);
    bits |= (unsigned)(v.w != 0) << (q * 4 + 3);
  }
  if ((row >> 5) == w) bits |= 1u << (row & 31);     // self loop
  mask[g] = bits;
}

// ---------------------------------------------------------------------------
// Kernel 2: Wh = x @ W ; Wh1 = Wh @ a[:16] ; Wh2 = Wh @ a[16:]
// Emits Wh directly in WMMA-B "fragment-ready" f16 layout:
//   lane L of chunk c reads v16h at frag[(c*32 + L)*16 .. +15], where
//   element e corresponds to memory row r = c*32 + (L>>4)*16 + e, feature L&15.
// One thread per node row.
// ---------------------------------------------------------------------------
__global__ void feat_kernel(const float* __restrict__ x,
                            const float* __restrict__ W,
                            const float* __restrict__ a,
                            _Float16* __restrict__ whfrag,
                            float* __restrict__ Wh1,
                            float* __restrict__ Wh2) {
  __shared__ float Ws[GF * GF];
  __shared__ float as[2 * GF];
  int tid = threadIdx.x;
  if (tid < GF * GF) Ws[tid] = W[tid];
  if (tid < 2 * GF)  as[tid] = a[tid];
  __syncthreads();

  int r = blockIdx.x * blockDim.x + tid;
  float xr[GF];
#pragma unroll
  for (int k = 0; k < GF; ++k) xr[k] = x[(size_t)r * GF + k];

  int chunk = r >> 5;
  int lh    = (r >> 4) & 1;
  int e     = r & 15;
  _Float16* fr = whfrag + ((size_t)chunk * 32 + lh * 16) * 16 + e;

  float w1 = 0.f, w2 = 0.f;
#pragma unroll
  for (int f = 0; f < GF; ++f) {
    float acc = 0.f;
#pragma unroll
    for (int k = 0; k < GF; ++k) acc = fmaf(xr[k], Ws[k * GF + f], acc);
    fr[(size_t)f * 16] = (_Float16)acc;      // lane = lh*16 + f
    w1 = fmaf(acc, as[f],      w1);
    w2 = fmaf(acc, as[GF + f], w2);
  }
  Wh1[r] = w1;
  Wh2[r] = w2;
}

// ---------------------------------------------------------------------------
// Kernel 3: per-row softmax statistics (max, sum of exp) over masked columns.
// One wave32 per row. Branchless: uniform mask-word broadcast, lane-per-column
// (conflict-free LDS reads of staged Wh2), cndmask selects instead of branches.
// ---------------------------------------------------------------------------
__global__ void stats_kernel(const unsigned* __restrict__ mask,
                             const float* __restrict__ Wh1,
                             const float* __restrict__ Wh2,
                             float* __restrict__ rmax,
                             float* __restrict__ rsum) {
  __shared__ float lwh2[GN];
  int tid = threadIdx.x;
  {
    const float4* src = (const float4*)Wh2;
    float4* dst = (float4*)lwh2;
    for (int i = tid; i < GN / 4; i += 256) dst[i] = src[i];
  }
  __syncthreads();

  int lane = tid & 31;
  int wave = tid >> 5;
  int row  = blockIdx.x * 8 + wave;
  float w1 = Wh1[row];
  const unsigned* mrow = mask + (size_t)row * GNW;

  float m = -1e30f;
  for (int g = 0; g < GNW; ++g) {
    unsigned bits = mrow[g];                 // same addr for all lanes
    float e = w1 + lwh2[g * 32 + lane];      // lane-consecutive, no conflicts
    e = e > 0.f ? e : GALPHA * e;
    m = ((bits >> lane) & 1u) ? fmaxf(m, e) : m;
  }
  for (int off = 16; off; off >>= 1) m = fmaxf(m, __shfl_xor(m, off));

  float s = 0.f;
  for (int g = 0; g < GNW; ++g) {
    unsigned bits = mrow[g];
    float e = w1 + lwh2[g * 32 + lane];
    e = e > 0.f ? e : GALPHA * e;
    float p = __expf(e - m);
    s += ((bits >> lane) & 1u) ? p : 0.f;
  }
  for (int off = 16; off; off >>= 1) s += __shfl_xor(s, off);

  if (lane == 0) { rmax[row] = m; rsum[row] = s; }
}

// ---------------------------------------------------------------------------
// Kernel 4: h' = softmax(masked e) @ Wh, then ELU.  Flash-style with WMMA.
// One block (4 waves) per 16-row tile; 4-way K-split (64 chunks per wave).
// Per chunk: branchless 16x32 f16 P-tile in a contiguous-K A-fragment layout
// (Wh2 from LDS, exp always computed, mask applied by select), one contiguous
// 32-byte fragment-ready B load, one v_wmma_f32_16x16x32_f16. Partial C
// fragments combined through padded LDS; wave 0 applies ELU and stores.
// ---------------------------------------------------------------------------
__global__ void gat_attn_wmma_kernel(const unsigned* __restrict__ mask,
                                     const float* __restrict__ Wh1,
                                     const float* __restrict__ Wh2,
                                     const float* __restrict__ rmax,
                                     const float* __restrict__ rsum,
                                     const _Float16* __restrict__ whfrag,
                                     float* __restrict__ out) {
  __shared__ float lwh2[GN];                 // 32 KB staged Wh2
  __shared__ float red[4][32][9];            // padded partial-C exchange
  int tid = threadIdx.x;
  {
    const float4* src = (const float4*)Wh2;
    float4* dst = (float4*)lwh2;
    for (int i = tid; i < GN / 4; i += 128) dst[i] = src[i];
  }
  __syncthreads();

  int lane = tid & 31;
  int wave = tid >> 5;
  int tile = blockIdx.x;

  int ma  = lane & 15;                       // A row within tile / D column
  int lh  = lane >> 4;                       // lane half selects K sub-block
  int row = tile * 16 + ma;

  float w1  = Wh1[row];
  float rm  = rmax[row];
  float inv = 1.0f / rsum[row];
  const unsigned* mrow = mask + (size_t)row * GNW;
  const v16h* bfrag = (const v16h*)whfrag;

  // K index inside the chunk for each of the 16 f16 A-fragment elements
  // (contiguous K order matching whfrag B layout: K = lh*16 + e)
  int kidx[16];
#pragma unroll
  for (int e = 0; e < 16; ++e) {
    kidx[e] = lh * 16 + e;
  }

  v8f c = {};
  int cbeg = wave * (GNW / 4);
  int cend = cbeg + (GNW / 4);
  for (int chunk = cbeg; chunk < cend; ++chunk) {
    unsigned wbits = mrow[chunk];
    __builtin_prefetch(mrow + chunk + 8, 0, 1);   // global_prefetch_b8
    int c0 = chunk * 32;

    v16h a;
#pragma unroll
    for (int e = 0; e < 16; ++e) {
      int kk = kidx[e];
      float s = w1 + lwh2[c0 + kk];               // LDS broadcast per half
      s = s > 0.f ? s : GALPHA * s;               // LeakyReLU
      float p = __expf(s - rm) * inv;             // always computed
      p = ((wbits >> kk) & 1u) ? p : 0.0f;        // mask via cndmask
      a[e] = (_Float16)p;
    }

    v16h b = bfrag[(size_t)chunk * 32 + lane];    // contiguous 32B load

    c = __builtin_amdgcn_wmma_f32_16x16x32_f16(
        /*neg_a=*/false, a, /*neg_b=*/false, b,
        /*c_mod=*/(short)0, c, /*reuse_a=*/false, /*reuse_b=*/false);
  }

#pragma unroll
  for (int r = 0; r < 8; ++r) red[wave][lane][r] = c[r];
  __syncthreads();

  if (wave == 0) {
#pragma unroll
    for (int r = 0; r < 8; ++r) {
      float v = red[0][lane][r] + red[1][lane][r] +
                red[2][lane][r] + red[3][lane][r];
      v = v > 0.f ? v : (__expf(v) - 1.0f);       // ELU
      int rowi = tile * 16 + (lh ? (8 + r) : r);  // D layout
      out[(size_t)rowi * GF + ma] = v;
    }
  }
}

// ---------------------------------------------------------------------------
// Host-side launch
// ---------------------------------------------------------------------------
extern "C" void kernel_launch(void* const* d_in, const int* in_sizes, int n_in,
                              void* d_out, int out_size, void* d_ws, size_t ws_size,
                              hipStream_t stream) {
  const float* h   = (const float*)d_in[0];
  const int*   adj = (const int*)  d_in[1];
  const float* W0  = (const float*)d_in[2];
  const float* a0  = (const float*)d_in[3];
  const float* W1  = (const float*)d_in[4];
  const float* a1  = (const float*)d_in[5];
  float* out = (float*)d_out;

  char* ws = (char*)d_ws;
  unsigned* mask   = (unsigned*)(ws + 0);          // 8 MB
  float*    x1     = (float*)   (ws + 8388608);    // 512 KB
  _Float16* whfrag = (_Float16*)(ws + 8912896);    // 256 KB
  float*    Wh1    = (float*)   (ws + 9175040);    // 32 KB
  float*    Wh2    = (float*)   (ws + 9207808);    // 32 KB
  float*    rmaxp  = (float*)   (ws + 9240576);    // 32 KB
  float*    rsump  = (float*)   (ws + 9273344);    // 32 KB

  // one-time adjacency -> bitmask compression (the 256MB scan)
  pack_mask_kernel<<<(GN * GNW) / 256, 256, 0, stream>>>(adj, mask);

  // ---- layer 0 ----
  feat_kernel<<<GN / 256, 256, 0, stream>>>(h, W0, a0, whfrag, Wh1, Wh2);
  stats_kernel<<<GN / 8, 256, 0, stream>>>(mask, Wh1, Wh2, rmaxp, rsump);
  gat_attn_wmma_kernel<<<GN / 16, 128, 0, stream>>>(mask, Wh1, Wh2,
                                                    rmaxp, rsump, whfrag, x1);

  // ---- layer 1 ----
  feat_kernel<<<GN / 256, 256, 0, stream>>>(x1, W1, a1, whfrag, Wh1, Wh2);
  stats_kernel<<<GN / 8, 256, 0, stream>>>(mask, Wh1, Wh2, rmaxp, rsump);
  gat_attn_wmma_kernel<<<GN / 16, 128, 0, stream>>>(mask, Wh1, Wh2,
                                                    rmaxp, rsump, whfrag, out);
}
